// UResidualMambaSeq_68564857913692
// MI455X (gfx1250) — matrix operand order repaired and checked by reading, not compiled, not run on hardware
//
#include <hip/hip_runtime.h>
#include <hip/hip_bf16.h>
#include <cmath>

// ---------------------------------------------------------------------------
// MI455X (gfx1250) implementation of UResidualMambaSeq forward.
// All dense contractions -> v_wmma_f32_16x16x32_f16 (f16 in, f32 accum),
// 64x64 block tiles, 4 waves, each wave = 2x2 grid of 16x16 WMMA accumulators
// (4 v_wmma per wave per 32-K step with A/B fragment reuse).
// Scans / elementwise in fp32. Reversal of the bwd branch is folded into
// direction flags of the depthwise conv and the scan (GEMMs commute with it).
// Workspace requirement: ~43 MB.
// ---------------------------------------------------------------------------

typedef __attribute__((ext_vector_type(16))) _Float16 v16h;
typedef __attribute__((ext_vector_type(8)))  _Float16 h8;
typedef __attribute__((ext_vector_type(8)))  float    v8f;

static inline int cdiv(int a, int b) { return (a + b - 1) / b; }

// ---------------------------------------------------------------------------
// WMMA GEMM: C[b] = act( op(A[b]) * op(B[b]) + bias ), templated on layout.
//   op(A): TA ? A[K][M] : A[M][K]
//   op(B): TB ? B[N][K] : B[K][N]
//   biasMode: 0 none, 1 bias[row(M)], 2 bias[col(N)]; act: 0 none,1 silu,2 softplus
// Block: 128 threads = 4 waves in 2x2; block tile 64x64, wave tile 32x32.
// f32 tiles staged to LDS as f16; fragments per CDNA5 ISA layouts:
//   A 16x32: lane m = lid&15, half h = lid>>4 -> K chunks [8h..8h+7],[16+8h..+7]
//   B 32x16: staged transposed (Bs[n][k]) so a lane's K-column is contiguous.
// ---------------------------------------------------------------------------
template <int TA, int TB>
__global__ __launch_bounds__(128) void gemm_wmma_kernel(
    const float* __restrict__ A, const float* __restrict__ Bm,
    const float* __restrict__ bias, float* __restrict__ C,
    int M, int N, int K, int lda, int ldb, int ldc,
    long sA, long sB, long sC, int biasMode, int act)
{
  __shared__ __align__(16) _Float16 As[64][40];   // [m][k], pitch 80B
  __shared__ __align__(16) _Float16 Bs[64][40];   // [n][k]

  const int tid  = threadIdx.x;
  const float* Ab = A  + sA * blockIdx.z;
  const float* Bb = Bm + sB * blockIdx.z;
  float*       Cb = C  + sC * blockIdx.z;
  const int m0 = blockIdx.y * 64, n0 = blockIdx.x * 64;
  const int lid  = tid & 31, wave = tid >> 5;
  const int wm   = wave >> 1, wn = wave & 1;     // 2x2 waves over 64x64
  const int mrow = lid & 15,  hf = lid >> 4;

  v8f acc00 = {}, acc01 = {}, acc10 = {}, acc11 = {};

  for (int kt = 0; kt < K; kt += 32) {
    if (kt) __syncthreads();
    // ---- stage 64x32 of A and B (f32 -> f16), zero-padded at edges ----
    #pragma unroll
    for (int e = 0; e < 16; e++) {
      int flat = e * 128 + tid;
      int r = flat >> 5, c = flat & 31;          // r: tile row (m or n), c: k
      int kk = kt + c;
      float av = 0.f, bv = 0.f;
      int mm = m0 + r;
      if (mm < M && kk < K) av = TA ? Ab[kk * lda + mm] : Ab[mm * lda + kk];
      int nn = n0 + r;
      if (nn < N && kk < K) bv = TB ? Bb[nn * ldb + kk] : Bb[kk * ldb + nn];
      As[r][c] = (_Float16)av;
      Bs[r][c] = (_Float16)bv;
    }
    // speculative prefetch of next K tile (global_prefetch_b8)
    if (kt + 32 < K) {
      int r = tid >> 1, c = (tid & 1) * 16;
      int kk = kt + 32 + c;
      if (m0 + r < M && kk < K)
        __builtin_prefetch(TA ? &Ab[kk * lda + m0 + r] : &Ab[(m0 + r) * lda + kk], 0, 1);
      if (n0 + r < N && kk < K)
        __builtin_prefetch(TB ? &Bb[(n0 + r) * ldb + kk] : &Bb[kk * ldb + n0 + r], 0, 1);
    }
    __syncthreads();

    // ---- fragments: 2 A-frags x 2 B-frags -> 4 WMMA ----
    const _Float16* ar0 = &As[32 * wm + mrow][0];
    const _Float16* ar1 = &As[32 * wm + 16 + mrow][0];
    const _Float16* br0 = &Bs[32 * wn + mrow][0];
    const _Float16* br1 = &Bs[32 * wn + 16 + mrow][0];
    union { v16h v; h8 h[2]; } a0, a1, b0, b1;
    a0.h[0] = *(const h8*)(ar0 + 8 * hf);  a0.h[1] = *(const h8*)(ar0 + 16 + 8 * hf);
    a1.h[0] = *(const h8*)(ar1 + 8 * hf);  a1.h[1] = *(const h8*)(ar1 + 16 + 8 * hf);
    b0.h[0] = *(const h8*)(br0 + 8 * hf);  b0.h[1] = *(const h8*)(br0 + 16 + 8 * hf);
    b1.h[0] = *(const h8*)(br1 + 8 * hf);  b1.h[1] = *(const h8*)(br1 + 16 + 8 * hf);

    acc00 = __builtin_amdgcn_wmma_f32_16x16x32_f16(false, a0.v, false, b0.v, (short)0, acc00, false, false);
    acc01 = __builtin_amdgcn_wmma_f32_16x16x32_f16(false, a0.v, false, b1.v, (short)0, acc01, false, false);
    acc10 = __builtin_amdgcn_wmma_f32_16x16x32_f16(false, a1.v, false, b0.v, (short)0, acc10, false, false);
    acc11 = __builtin_amdgcn_wmma_f32_16x16x32_f16(false, a1.v, false, b1.v, (short)0, acc11, false, false);
  }

  // ---- epilogue: C layout: vgpr r -> M = r + 8*half, N = lane&15 ----
  #pragma unroll
  for (int i = 0; i < 2; i++) {
    #pragma unroll
    for (int j = 0; j < 2; j++) {
      v8f acc = (i == 0) ? (j == 0 ? acc00 : acc01) : (j == 0 ? acc10 : acc11);
      int rb = m0 + 32 * wm + 16 * i + 8 * hf;
      int col = n0 + 32 * wn + 16 * j + mrow;
      #pragma unroll
      for (int r2 = 0; r2 < 8; r2++) {
        int row = rb + r2;
        if (row < M && col < N) {
          float v = acc[r2];
          if (biasMode == 1) v += bias[row];
          else if (biasMode == 2) v += bias[col];
          if (act == 1) v = v / (1.f + __expf(-v));                   // silu
          else if (act == 2) v = (v > 20.f) ? v : log1pf(__expf(v));  // softplus
          Cb[row * ldc + col] = v;
        }
      }
    }
  }
}

// ---------------------------------------------------------------------------
// RMSNorm over contiguous rows of length n; optional residual add.
// ---------------------------------------------------------------------------
__global__ __launch_bounds__(256) void rmsnorm_kernel(
    const float* __restrict__ x, const float* __restrict__ res,
    float* __restrict__ out, int n)
{
  __shared__ float red[256];
  long row = blockIdx.x;
  const float* xr = x + row * n;
  float s = 0.f;
  for (int i = threadIdx.x; i < n; i += 256) { float v = xr[i]; s += v * v; }
  red[threadIdx.x] = s;
  __syncthreads();
  for (int off = 128; off; off >>= 1) {
    if (threadIdx.x < off) red[threadIdx.x] += red[threadIdx.x + off];
    __syncthreads();
  }
  float scale = rsqrtf(red[0] / (float)n + 1e-8f);
  for (int i = threadIdx.x; i < n; i += 256) {
    float v = xr[i] * scale;
    if (res) v += res[row * n + i];
    out[row * n + i] = v;
  }
}

// ---------------------------------------------------------------------------
// Depthwise causal conv (D_CONV=4) + SiLU. u0 layout (b, l, 2di) (reads first
// di columns), output u (b, l, di). dir=0 causal, dir=1 anti-causal (flipped).
// ---------------------------------------------------------------------------
__global__ __launch_bounds__(256) void dwconv_silu_kernel(
    const float* __restrict__ u0, const float* __restrict__ wgt,
    const float* __restrict__ bias, float* __restrict__ u,
    int di, int ld, int L, int dir, int total)
{
  int idx = blockIdx.x * 256 + threadIdx.x;
  if (idx >= total) return;
  int d = idx % di;
  int l = (idx / di) % L;
  int b = idx / (di * L);
  const float* base = u0 + (long)b * L * ld + d;
  float acc = bias[d];
  #pragma unroll
  for (int j = 0; j < 4; j++) {
    int src = dir ? (l + 3 - j) : (l - 3 + j);
    if (src >= 0 && src < L) acc += wgt[d * 4 + j] * base[(long)src * ld];
  }
  acc = acc / (1.f + __expf(-acc));
  u[idx] = acc;
}

// ---------------------------------------------------------------------------
// Selective scan: one thread per (b, d); 16 states in registers; L steps.
// ---------------------------------------------------------------------------
__global__ __launch_bounds__(256) void scan_kernel(
    const float* __restrict__ dt, const float* __restrict__ u,
    const float* __restrict__ xdbl, const float* __restrict__ Alog,
    const float* __restrict__ Dp, float* __restrict__ y,
    int di, int r, int L, int dir, int total)
{
  int idx = blockIdx.x * 256 + threadIdx.x;
  if (idx >= total) return;
  int d = idx % di, b = idx / di;
  int ldx = r + 32;
  float An[16], h[16];
  #pragma unroll
  for (int n = 0; n < 16; n++) { An[n] = -__expf(Alog[d * 16 + n]); h[n] = 0.f; }
  float Dd = Dp[d];
  for (int t = 0; t < L; t++) {
    int l = dir ? (L - 1 - t) : t;
    long row = (long)b * L + l;
    float dtv = dt[row * di + d];
    float uv  = u[row * di + d];
    const float* xr = xdbl + row * ldx + r;   // B at [0..15], C at [16..31]
    float acc = uv * Dd;
    float dtu = dtv * uv;
    #pragma unroll
    for (int n = 0; n < 16; n++) {
      float dA = __expf(dtv * An[n]);
      h[n] = dA * h[n] + dtu * xr[n];
      acc += h[n] * xr[16 + n];
    }
    y[row * di + d] = acc;
  }
}

// y *= silu(z), z read at (row*ldz + d) (second half of the in_proj output).
__global__ __launch_bounds__(256) void mul_silu_kernel(
    float* __restrict__ y, const float* __restrict__ z,
    int ldz, int di, int total)
{
  int idx = blockIdx.x * 256 + threadIdx.x;
  if (idx >= total) return;
  int d = idx % di; long row = idx / di;
  float zv = z[row * ldz + d];
  y[idx] *= zv / (1.f + __expf(-zv));
}

// comb[b,o,c] = w[b,o,c] * (pm_fwd[b,c,o] + pm_bwd[b,c,o])   (w already silu'd)
__global__ __launch_bounds__(256) void combine_kernel(
    const float* __restrict__ wv, const float* __restrict__ pf,
    const float* __restrict__ pb, float* __restrict__ out, int twoL, int total)
{
  int idx = blockIdx.x * 256 + threadIdx.x;
  if (idx >= total) return;
  int c = idx % 256;
  int o = (idx / 256) % twoL;
  int b = idx / (256 * twoL);
  long p = ((long)b * 256 + c) * twoL + o;
  out[idx] = wv[idx] * (pf[p] + pb[p]);
}

// out[b,c,s] = res[b,c,s] + enc[b,s,c]
__global__ __launch_bounds__(256) void add_transpose_kernel(
    const float* __restrict__ enc, const float* __restrict__ res,
    float* __restrict__ out, int L, int total)
{
  int idx = blockIdx.x * 256 + threadIdx.x;
  if (idx >= total) return;
  int s = idx % L;
  int c = (idx / L) % 256;
  int b = idx / (L * 256);
  out[idx] = res[idx] + enc[((long)b * L + s) * 256 + c];
}

__global__ __launch_bounds__(256) void add_kernel(
    const float* __restrict__ a, const float* __restrict__ b,
    float* __restrict__ o, int n)
{
  int i = blockIdx.x * 256 + threadIdx.x;
  if (i < n) o[i] = a[i] + b[i];
}

// ---------------------------------------------------------------------------
// Host-side orchestration
// ---------------------------------------------------------------------------
struct Conv1x1P { const float* w; const float* b; };
struct MambaP {
  const float *in_proj, *conv_w, *conv_b, *x_proj, *dt_w, *dt_b, *A_log, *D, *out_proj;
};
struct EncP {
  Conv1x1P input_proj, weight_proj, fwd_proj, bwd_proj, out_proj;
  MambaP fwd, bwd;
};
struct Ws {
  float *xn, *ssm, *wbuf, *brx0, *brx1, *pm0, *pm1, *mout, *comb, *encout;
  float *xz, *u, *xdbl, *dt, *y;
};

static Conv1x1P parseConv(void* const* din, int& c) {
  Conv1x1P p; p.w = (const float*)din[c++]; p.b = (const float*)din[c++]; return p;
}
static MambaP parseMamba(void* const* din, int& c) {
  MambaP m;
  m.in_proj = (const float*)din[c++]; m.conv_w = (const float*)din[c++];
  m.conv_b  = (const float*)din[c++]; m.x_proj = (const float*)din[c++];
  m.dt_w    = (const float*)din[c++]; m.dt_b   = (const float*)din[c++];
  m.A_log   = (const float*)din[c++]; m.D      = (const float*)din[c++];
  m.out_proj = (const float*)din[c++];
  return m;
}
static EncP parseEnc(void* const* din, int& c) {
  EncP e;
  e.input_proj  = parseConv(din, c);
  e.weight_proj = parseConv(din, c);
  e.fwd_proj    = parseConv(din, c);
  e.bwd_proj    = parseConv(din, c);
  e.out_proj    = parseConv(din, c);
  e.fwd = parseMamba(din, c);
  e.bwd = parseMamba(din, c);
  return e;
}

static inline void gemm(hipStream_t st,
                        const float* A, long sA, int lda, int tA,
                        const float* B, long sB, int ldb, int tB,
                        const float* bias, int biasMode, int act,
                        float* C, long sC, int ldc,
                        int M, int N, int K, int batch)
{
  dim3 g(cdiv(N, 64), cdiv(M, 64), batch), blk(128);
  if (!tA && !tB)
    gemm_wmma_kernel<0, 0><<<g, blk, 0, st>>>(A, B, bias, C, M, N, K, lda, ldb, ldc, sA, sB, sC, biasMode, act);
  else if (!tA && tB)
    gemm_wmma_kernel<0, 1><<<g, blk, 0, st>>>(A, B, bias, C, M, N, K, lda, ldb, ldc, sA, sB, sC, biasMode, act);
  else if (tA && !tB)
    gemm_wmma_kernel<1, 0><<<g, blk, 0, st>>>(A, B, bias, C, M, N, K, lda, ldb, ldc, sA, sB, sC, biasMode, act);
  else
    gemm_wmma_kernel<1, 1><<<g, blk, 0, st>>>(A, B, bias, C, M, N, K, lda, ldb, ldc, sA, sB, sC, biasMode, act);
}

// mamba on xin (B,256,dm) -> mout (B,256,dm). dir=1 => reversed-sequence branch.
static void run_mamba(hipStream_t st, const MambaP& m, const float* xin,
                      int dm, float* mout, int dir, const Ws& w)
{
  const int B = 4, L = 256;
  const int di = 2 * dm, r = dm / 16;
  // xz = xin * in_proj^T  (B,256,2di)
  gemm(st, xin, (long)L * dm, dm, 0, m.in_proj, 0, dm, 1,
       nullptr, 0, 0, w.xz, (long)L * 2 * di, 2 * di, L, 2 * di, dm, B);
  // u = silu(depthwise_conv(xz[..., :di]))
  int tot = B * L * di;
  dwconv_silu_kernel<<<cdiv(tot, 256), 256, 0, st>>>(
      w.xz, m.conv_w, m.conv_b, w.u, di, 2 * di, L, dir, tot);
  // x_dbl = u * x_proj^T  (B,256,r+32)
  gemm(st, w.u, (long)L * di, di, 0, m.x_proj, 0, di, 1,
       nullptr, 0, 0, w.xdbl, (long)L * (r + 32), r + 32, L, r + 32, di, B);
  // dt = softplus(x_dbl[:, :, :r] * dt_w^T + dt_b)  (B,256,di)
  gemm(st, w.xdbl, (long)L * (r + 32), r + 32, 0, m.dt_w, 0, r, 1,
       m.dt_b, 2, 2, w.dt, (long)L * di, di, L, di, r, B);
  // selective scan
  int tt = B * di;
  scan_kernel<<<cdiv(tt, 256), 256, 0, st>>>(
      w.dt, w.u, w.xdbl, m.A_log, m.D, w.y, di, r, L, dir, tt);
  // y *= silu(z)
  mul_silu_kernel<<<cdiv(tot, 256), 256, 0, st>>>(w.y, w.xz + di, 2 * di, di, tot);
  // mout = y * out_proj^T
  gemm(st, w.y, (long)L * di, di, 0, m.out_proj, 0, di, 1,
       nullptr, 0, 0, mout, (long)L * dm, dm, L, dm, di, B);
}

// encoder in-place on x (B,256,L)
static void run_encoder(hipStream_t st, const EncP& e, float* x, int L, const Ws& w)
{
  const int B = 4, C = 256;
  const int twoL = 2 * L, dm = twoL;
  // xn = rmsnorm(x) over L; rows = B*C (contiguous)
  rmsnorm_kernel<<<B * C, 256, 0, st>>>(x, nullptr, w.xn, L);
  // ssm[b,o,c] = Wip[o,s] * xn[b,c,s] + bip[o] : M=2L,N=C,K=L (B = xn, transB)
  gemm(st, e.input_proj.w, 0, L, 0, w.xn, (long)C * L, L, 1,
       e.input_proj.b, 1, 0, w.ssm, (long)twoL * C, C, twoL, C, L, B);
  // w[b,o,c] = silu(Wwp * xn + bwp)
  gemm(st, e.weight_proj.w, 0, L, 0, w.xn, (long)C * L, L, 1,
       e.weight_proj.b, 1, 1, w.wbuf, (long)twoL * C, C, twoL, C, L, B);
  // fwd_x[b,c,o'] = ssm[b,o,c] * Wfp[o',o] + bfp[o'] : transA(ssm), transB(Wfp)
  gemm(st, w.ssm, (long)twoL * C, C, 1, e.fwd_proj.w, 0, twoL, 1,
       e.fwd_proj.b, 2, 0, w.brx0, (long)C * twoL, twoL, C, twoL, twoL, B);
  gemm(st, w.ssm, (long)twoL * C, C, 1, e.bwd_proj.w, 0, twoL, 1,
       e.bwd_proj.b, 2, 0, w.brx1, (long)C * twoL, twoL, C, twoL, twoL, B);
  // postnorm mamba, fwd then bwd (bwd = dir-flagged conv + scan, no reversal)
  run_mamba(st, e.fwd, w.brx0, dm, w.mout, 0, w);
  rmsnorm_kernel<<<B * C, 256, 0, st>>>(w.mout, w.brx0, w.pm0, dm);
  run_mamba(st, e.bwd, w.brx1, dm, w.mout, 1, w);
  rmsnorm_kernel<<<B * C, 256, 0, st>>>(w.mout, w.brx1, w.pm1, dm);
  // comb = w * (pm_fwd^T + pm_bwd^T)
  int tot = B * twoL * C;
  combine_kernel<<<cdiv(tot, 256), 256, 0, st>>>(w.wbuf, w.pm0, w.pm1, w.comb, twoL, tot);
  // enc_out[b,m,c] = Wop[m,o] * comb[b,o,c] + bop[m]
  gemm(st, e.out_proj.w, 0, twoL, 0, w.comb, (long)twoL * C, C, 0,
       e.out_proj.b, 1, 0, w.encout, (long)L * C, C, L, C, twoL, B);
  // x = enc_out^T + x
  int tot2 = B * C * L;
  add_transpose_kernel<<<cdiv(tot2, 256), 256, 0, st>>>(w.encout, x, x, L, tot2);
}

extern "C" void kernel_launch(void* const* d_in, const int* in_sizes, int n_in,
                              void* d_out, int out_size, void* d_ws, size_t ws_size,
                              hipStream_t stream)
{
  (void)in_sizes; (void)n_in; (void)out_size; (void)ws_size;
  const int B = 4;

  // ---- parameter unpacking (dict order of setup_inputs) ----
  int cur = 0;
  const float* x_in = (const float*)d_in[cur++];
  Conv1x1P dconv[2]; EncP denc[2];
  for (int i = 0; i < 2; i++) { dconv[i] = parseConv(d_in, cur); denc[i] = parseEnc(d_in, cur); }
  EncP mide[2];
  for (int j = 0; j < 2; j++) mide[j] = parseEnc(d_in, cur);
  Conv1x1P uconv[2]; EncP uenc[2];
  for (int j = 0; j < 2; j++) { uconv[j] = parseConv(d_in, cur); uenc[j] = parseEnc(d_in, cur); }

  // ---- workspace slots (floats); total ~10.6M floats (~43 MB) ----
  float* p = (float*)d_ws;
  auto alloc = [&](size_t n) { float* r = p; p += n; return r; };
  float* x1 = alloc(262144);   // (4,256,256) max
  float* x0 = alloc(262144);
  float* x2 = alloc(65536);    // (4,256,64)
  Ws w;
  w.xn     = alloc(262144);
  w.ssm    = alloc(524288);
  w.wbuf   = alloc(524288);
  w.brx0   = alloc(524288);
  w.brx1   = alloc(524288);
  w.pm0    = alloc(524288);
  w.pm1    = alloc(524288);
  w.mout   = alloc(524288);
  w.comb   = alloc(524288);
  w.encout = alloc(262144);
  w.xz     = alloc(2097152);   // (4,256,2048) max
  w.u      = alloc(1048576);
  w.xdbl   = alloc(65536);
  w.dt     = alloc(1048576);
  w.y      = alloc(1048576);

  // ---- down path ----
  // level 0: seq 256 -> 128 (dense over seq axis), then encoder(L=128)
  gemm(stream, x_in, 256L * 256, 256, 0, dconv[0].w, 0, 256, 1,
       dconv[0].b, 2, 0, x1, 256L * 128, 128, 256, 128, 256, B);
  run_encoder(stream, denc[0], x1, 128, w);
  // level 1: 128 -> 64, encoder(L=64)
  gemm(stream, x1, 256L * 128, 128, 0, dconv[1].w, 0, 128, 1,
       dconv[1].b, 2, 0, x2, 256L * 64, 64, 256, 64, 128, B);
  run_encoder(stream, denc[1], x2, 64, w);

  // ---- mid ----
  run_encoder(stream, mide[0], x2, 64, w);
  run_encoder(stream, mide[1], x2, 64, w);

  // ---- up path ----
  // up 0: 64 -> 128, encoder(L=128), + residual (x1, untouched since down-0)
  gemm(stream, x2, 256L * 64, 64, 0, uconv[0].w, 0, 64, 1,
       uconv[0].b, 2, 0, x0, 256L * 128, 128, 256, 128, 64, B);
  run_encoder(stream, uenc[0], x0, 128, w);
  add_kernel<<<cdiv(131072, 256), 256, 0, stream>>>(x0, x1, x0, 131072);
  // up 1: 128 -> 256, encoder(L=256), + residual (original input), -> d_out
  gemm(stream, x0, 256L * 128, 128, 0, uconv[1].w, 0, 128, 1,
       uconv[1].b, 2, 0, x1, 256L * 256, 256, 256, 256, 128, B);
  run_encoder(stream, uenc[1], x1, 256, w);
  add_kernel<<<cdiv(262144, 256), 256, 0, stream>>>(x1, x_in, (float*)d_out, 262144);
}
